// NestedFeedForward_73761768341873
// MI455X (gfx1250) — matrix operand order repaired and checked by reading, not compiled, not run on hardware
//
#include <hip/hip_runtime.h>

#define DIM   1024
#define HID   4096
#define NEXP  4
#define NTOK  16384
#define TTILE 32
#define HPAD  (HID + 8)   // +16B per row: bank-conflict-free lane-strided LDS b128 loads

typedef __attribute__((ext_vector_type(16))) __bf16 v16bf;
typedef __attribute__((ext_vector_type(8)))  float  v8f;

union FragU { uint4 q[2]; v16bf v; };

__device__ __forceinline__ __bf16 f2bf(float f) {
  union { float f; unsigned u; } a; a.f = f;
  unsigned u = a.u;
  unsigned r = u + 0x7FFFu + ((u >> 16) & 1u);   // round-to-nearest-even
  union { unsigned short s; __bf16 b; } o;
  o.s = (unsigned short)(r >> 16);
  return o.b;
}

__device__ __forceinline__ float geluf(float x) {
  return 0.5f * x * (1.0f + erff(x * 0.70710678118654752f));  // exact erf GELU
}

// A-frag 16x32 bf16: lane holds row (lane&15), K = [ka,ka+8) U [ka+16,ka+24), ka=(lane>>4)*8
__device__ __forceinline__ v16bf ld_a(const __bf16* p) {
  FragU f;
  f.q[0] = *(const uint4*)p;
  f.q[1] = *(const uint4*)(p + 16);
  return f.v;
}
// B-frag 32x16 bf16: lane holds col (lane&15), K = [kb,kb+8) U [kb+16,kb+24), kb=(lane>>4)*8
__device__ __forceinline__ v16bf ld_b(const __bf16* p) {
  FragU f;
  f.q[0] = *(const uint4*)p;
  f.q[1] = *(const uint4*)(p + 16);
  return f.v;
}

// ---------------- conversion kernels ----------------

__global__ __launch_bounds__(256) void cvt_weights(
    const float4* __restrict__ w1, const float4* __restrict__ w2,
    uint2* __restrict__ w1b, uint2* __restrict__ w2b)
{
  const int n4 = (HID * DIM) / 4;     // 1048576 float4 per weight
  int i = blockIdx.x * blockDim.x + threadIdx.x;
  float4 v;
  uint2* dst;
  if (i < n4) { v = w1[i]; dst = w1b + i; }
  else        { v = w2[i - n4]; dst = w2b + (i - n4); }
  union { __bf16 b[4]; uint2 u; } o;
  o.b[0] = f2bf(v.x); o.b[1] = f2bf(v.y); o.b[2] = f2bf(v.z); o.b[3] = f2bf(v.w);
  *dst = o.u;
}

__global__ __launch_bounds__(256) void cvt_x(
    const float4* __restrict__ x, const int* __restrict__ mask,
    uint2* __restrict__ xb)
{
  const int t = blockIdx.x;
  const int d = DIM >> (NEXP - 1 - mask[t]);    // matryoshka input width
  const int i = threadIdx.x;                    // 256 threads * 4 floats = 1024
  const int k = i * 4;
  float4 v = x[t * (DIM / 4) + i];
  union { __bf16 b[4]; uint2 u; } o;
  o.b[0] = (k + 0 < d) ? f2bf(v.x) : f2bf(0.0f);
  o.b[1] = (k + 1 < d) ? f2bf(v.y) : f2bf(0.0f);
  o.b[2] = (k + 2 < d) ? f2bf(v.z) : f2bf(0.0f);
  o.b[3] = (k + 3 < d) ? f2bf(v.w) : f2bf(0.0f);
  xb[t * (DIM / 4) + i] = o.u;
}

// ---------------- fused nested FFN ----------------
// 1 workgroup = 32 tokens, 256 threads = 8 wave32s.
// GEMM1 (x@w1^T + b1 -> gelu) keeps h[32][4096] bf16 in LDS (257 KB, needs CDNA5's
// 320KB WGP LDS); GEMM2 reads it back. Each wave register-blocks 32(M) x 64(N):
// 2 A-frags x 4 B-frags -> 8 WMMA per 12 b128 loads.

__global__ __launch_bounds__(256, 1) void ffn_fused(
    const __bf16* __restrict__ xb,  const __bf16* __restrict__ w1b,
    const float*  __restrict__ b1,  const __bf16* __restrict__ w2b,
    const float*  __restrict__ b2,  const int* __restrict__ mask,
    float* __restrict__ out)
{
  __shared__ __bf16 hbuf[TTILE * HPAD];   // 32 * 4104 * 2B = 262656 B
  __shared__ int    dl[TTILE];

  const int t0   = blockIdx.x * TTILE;
  const int tid  = threadIdx.x;
  const int lane = tid & 31;
  const int wv   = tid >> 5;

  if (tid < TTILE) dl[tid] = DIM >> (NEXP - 1 - mask[t0 + tid]);
  __syncthreads();

  const int mr = lane & 15;          // A row / B col / C col
  const int ka = (lane >> 4) * 8;    // A k-chunk base
  const int kb = (lane >> 4) * 8;    // B k-chunk base
  const int mc = (lane >> 4) * 8;    // C row base

  const __bf16* xrow0 = xb + (size_t)(t0 + mr) * DIM + ka;
  const __bf16* xrow1 = xrow0 + (size_t)16 * DIM;

  // ---------- GEMM1: wave wv covers output cols [wv*512, wv*512+512) ----------
  for (int nb = 0; nb < 8; ++nb) {
    const int n0 = wv * 512 + nb * 64;
    v8f acc[8] = {v8f{}, v8f{}, v8f{}, v8f{}, v8f{}, v8f{}, v8f{}, v8f{}};
    const __bf16* wp0 = w1b + (size_t)(n0 +  0 + mr) * DIM + kb;
    const __bf16* wp1 = w1b + (size_t)(n0 + 16 + mr) * DIM + kb;
    const __bf16* wp2 = w1b + (size_t)(n0 + 32 + mr) * DIM + kb;
    const __bf16* wp3 = w1b + (size_t)(n0 + 48 + mr) * DIM + kb;
    for (int k = 0; k < DIM; k += 32) {
      v16bf a0 = ld_a(xrow0 + k);
      v16bf a1 = ld_a(xrow1 + k);
      v16bf f0 = ld_b(wp0 + k);
      v16bf f1 = ld_b(wp1 + k);
      v16bf f2 = ld_b(wp2 + k);
      v16bf f3 = ld_b(wp3 + k);
      acc[0] = __builtin_amdgcn_wmma_f32_16x16x32_bf16(false, a0, false, f0, (short)0, acc[0], false, false);
      acc[1] = __builtin_amdgcn_wmma_f32_16x16x32_bf16(false, a0, false, f1, (short)0, acc[1], false, false);
      acc[2] = __builtin_amdgcn_wmma_f32_16x16x32_bf16(false, a0, false, f2, (short)0, acc[2], false, false);
      acc[3] = __builtin_amdgcn_wmma_f32_16x16x32_bf16(false, a0, false, f3, (short)0, acc[3], false, false);
      acc[4] = __builtin_amdgcn_wmma_f32_16x16x32_bf16(false, a1, false, f0, (short)0, acc[4], false, false);
      acc[5] = __builtin_amdgcn_wmma_f32_16x16x32_bf16(false, a1, false, f1, (short)0, acc[5], false, false);
      acc[6] = __builtin_amdgcn_wmma_f32_16x16x32_bf16(false, a1, false, f2, (short)0, acc[6], false, false);
      acc[7] = __builtin_amdgcn_wmma_f32_16x16x32_bf16(false, a1, false, f3, (short)0, acc[7], false, false);
    }
#pragma unroll
    for (int mi = 0; mi < 2; ++mi) {
#pragma unroll
      for (int j = 0; j < 4; ++j) {
        const int n = n0 + j * 16 + mr;
        const float bias = b1[n];
#pragma unroll
        for (int r = 0; r < 8; ++r) {
          const int m = mi * 16 + mc + r;
          hbuf[m * HPAD + n] = f2bf(geluf(acc[mi * 4 + j][r] + bias));
        }
      }
    }
  }
  __syncthreads();

  // ---------- GEMM2: wave wv covers output cols [wv*128, wv*128+128) ----------
  for (int nb = 0; nb < 2; ++nb) {
    const int n0 = wv * 128 + nb * 64;
    v8f acc[8] = {v8f{}, v8f{}, v8f{}, v8f{}, v8f{}, v8f{}, v8f{}, v8f{}};
    const __bf16* wp0 = w2b + (size_t)(n0 +  0 + mr) * HID + kb;
    const __bf16* wp1 = w2b + (size_t)(n0 + 16 + mr) * HID + kb;
    const __bf16* wp2 = w2b + (size_t)(n0 + 32 + mr) * HID + kb;
    const __bf16* wp3 = w2b + (size_t)(n0 + 48 + mr) * HID + kb;
    for (int k = 0; k < HID; k += 32) {
      FragU fa0, fa1;
      const __bf16* hp0 = &hbuf[mr * HPAD + k + ka];
      const __bf16* hp1 = &hbuf[(16 + mr) * HPAD + k + ka];
      fa0.q[0] = *(const uint4*)hp0;
      fa0.q[1] = *(const uint4*)(hp0 + 16);
      fa1.q[0] = *(const uint4*)hp1;
      fa1.q[1] = *(const uint4*)(hp1 + 16);
      v16bf a0 = fa0.v;
      v16bf a1 = fa1.v;
      v16bf f0 = ld_b(wp0 + k);
      v16bf f1 = ld_b(wp1 + k);
      v16bf f2 = ld_b(wp2 + k);
      v16bf f3 = ld_b(wp3 + k);
      acc[0] = __builtin_amdgcn_wmma_f32_16x16x32_bf16(false, a0, false, f0, (short)0, acc[0], false, false);
      acc[1] = __builtin_amdgcn_wmma_f32_16x16x32_bf16(false, a0, false, f1, (short)0, acc[1], false, false);
      acc[2] = __builtin_amdgcn_wmma_f32_16x16x32_bf16(false, a0, false, f2, (short)0, acc[2], false, false);
      acc[3] = __builtin_amdgcn_wmma_f32_16x16x32_bf16(false, a0, false, f3, (short)0, acc[3], false, false);
      acc[4] = __builtin_amdgcn_wmma_f32_16x16x32_bf16(false, a1, false, f0, (short)0, acc[4], false, false);
      acc[5] = __builtin_amdgcn_wmma_f32_16x16x32_bf16(false, a1, false, f1, (short)0, acc[5], false, false);
      acc[6] = __builtin_amdgcn_wmma_f32_16x16x32_bf16(false, a1, false, f2, (short)0, acc[6], false, false);
      acc[7] = __builtin_amdgcn_wmma_f32_16x16x32_bf16(false, a1, false, f3, (short)0, acc[7], false, false);
    }
#pragma unroll
    for (int mi = 0; mi < 2; ++mi) {
#pragma unroll
      for (int j = 0; j < 4; ++j) {
        const int n = n0 + j * 16 + mr;
        const float bias = b2[n];
#pragma unroll
        for (int r = 0; r < 8; ++r) {
          const int m = mi * 16 + mc + r;
          float v = acc[mi * 4 + j][r] + bias;
          if (n >= dl[m]) v = 0.0f;               // matryoshka output truncation
          out[(size_t)(t0 + m) * DIM + n] = v;
        }
      }
    }
  }
}

// ---------------- launcher ----------------

extern "C" void kernel_launch(void* const* d_in, const int* in_sizes, int n_in,
                              void* d_out, int out_size, void* d_ws, size_t ws_size,
                              hipStream_t stream) {
  const float* x   = (const float*)d_in[0];
  const float* w1  = (const float*)d_in[1];
  const float* b1  = (const float*)d_in[2];
  const float* w2  = (const float*)d_in[3];
  const float* b2  = (const float*)d_in[4];
  const int*  mask = (const int*)d_in[5];
  float* out = (float*)d_out;

  char* ws = (char*)d_ws;
  __bf16* xb  = (__bf16*)ws;                                       // 32 MB
  __bf16* w1b = (__bf16*)(ws + (size_t)NTOK * DIM * 2);            //  8 MB
  __bf16* w2b = (__bf16*)(ws + (size_t)NTOK * DIM * 2
                             + (size_t)HID * DIM * 2);             //  8 MB

  // convert weights: 2 * 1048576 float4 -> 8192 blocks * 256
  cvt_weights<<<8192, 256, 0, stream>>>((const float4*)w1, (const float4*)w2,
                                        (uint2*)w1b, (uint2*)w2b);
  // convert + input-mask x: one block per token
  cvt_x<<<NTOK, 256, 0, stream>>>((const float4*)x, mask, (uint2*)xb);
  // fused FFN: one workgroup per 32-token tile
  ffn_fused<<<NTOK / TTILE, 256, 0, stream>>>(xb, w1b, b1, w2b, b2, mask, out);
}
